// LSTMLM_65231963292193
// MI455X (gfx1250) — compile-verified, matrix operand order checked by reading, not run on hardware
//
#include <hip/hip_runtime.h>
#include <hip/hip_bf16.h>

#define VOCAB 32000
#define DMODEL 1024
#define NLAYER 2
#define BATCH 8
#define SEQ 256
#define GATES 4096            // 4*DMODEL
#define MTOT 2048             // BATCH*SEQ

typedef __attribute__((ext_vector_type(16))) __bf16    v16bf;
typedef __attribute__((ext_vector_type(8)))  float     v8f;
typedef __attribute__((ext_vector_type(4)))  unsigned  u32x4;   // 128-bit, union-safe

union ABf { v16bf v; unsigned u[8]; u32x4 q[2]; };

__device__ __forceinline__ unsigned short f2bf(float f) {
    union { float f; unsigned u; } x; x.f = f;
    unsigned r = x.u + 0x7FFFu + ((x.u >> 16) & 1u);   // round-to-nearest-even
    return (unsigned short)(r >> 16);
}
__device__ __forceinline__ float bf2f(unsigned short h) {
    union { unsigned u; float f; } x; x.u = ((unsigned)h) << 16;
    return x.f;
}
__device__ __forceinline__ unsigned pack_bf2(float lo, float hi) {
    return ((unsigned)f2bf(hi) << 16) | (unsigned)f2bf(lo);
}

// ---------------------------------------------------------------------------
// Bulk fp32 -> bf16 conversion (packed u32 stores); n2 = element-pairs
// ---------------------------------------------------------------------------
__global__ __launch_bounds__(256)
void cvt_bf16(const float* __restrict__ src, unsigned* __restrict__ dst, long n2) {
    long i = (long)blockIdx.x * 256 + threadIdx.x;
    if (i < n2) dst[i] = pack_bf2(src[2 * i], src[2 * i + 1]);
}

// ---------------------------------------------------------------------------
// Embedding gather + fp32 -> bf16
// ---------------------------------------------------------------------------
__global__ __launch_bounds__(256)
void embed_bf16(const int* __restrict__ ids, const float* __restrict__ emb,
                unsigned* __restrict__ xbf_u32) {
    int row = blockIdx.x;                     // b*SEQ + s
    int id  = ids[row];
    const float* src = emb + (size_t)id * DMODEL;
    for (int j = threadIdx.x; j < DMODEL / 2; j += 256)
        xbf_u32[(size_t)row * (DMODEL / 2) + j] = pack_bf2(src[2 * j], src[2 * j + 1]);
}

// ---------------------------------------------------------------------------
// Tiled WMMA GEMM:  C[M,N] = A_bf16[M,1024] @ B_bf16[N,1024]^T (+bias1+bias2)
// 256 threads / 8 waves. WG tile 128x128; per-wave tile 32x64
// (2 M-frags x 4 N-frags -> 8 accumulators). K staged 32-wide in
// double-buffered LDS; staging uses b128 global loads issued before the
// b128 LDS stores so the loads pipeline under one descending loadcnt wait.
// ---------------------------------------------------------------------------
__global__ __launch_bounds__(256)
void gemm_bf16_wmma(const unsigned short* __restrict__ Abf,
                    const unsigned short* __restrict__ Bbf,
                    const float* __restrict__ bias1,
                    const float* __restrict__ bias2,
                    float* __restrict__ C, int ldc)
{
    __shared__ unsigned Asm[2][128 * 16];   // 128 rows x 32 bf16 (16 u32) x2 buf
    __shared__ unsigned Bsm[2][128 * 16];

    const int tid  = threadIdx.x;
    const int wave = tid >> 5, lane = tid & 31;
    const int l15  = lane & 15, hi = lane >> 4;
    const int wm   = wave & 3;          // M quadrant (32 rows)
    const int wn   = wave >> 2;         // N half (64 cols)
    const int n0   = blockIdx.x * 128, m0 = blockIdx.y * 128;
    const u32x4* Ag = (const u32x4*)Abf;    // row stride 128 u32x4
    const u32x4* Bg = (const u32x4*)Bbf;

    v8f acc[2][4];
#pragma unroll
    for (int mi = 0; mi < 2; ++mi)
#pragma unroll
        for (int ni = 0; ni < 4; ++ni) acc[mi][ni] = v8f{};

    // stage one 32-wide K chunk into LDS buffer `buf` (b128 traffic only)
    auto stage = [&](int buf, int k0) {
        u32x4* As4 = (u32x4*)Asm[buf];
        u32x4* Bs4 = (u32x4*)Bsm[buf];
        const int kq = k0 >> 3;                         // u32x4 offset within row
        const int vi0 = tid, vi1 = tid + 256;           // 512 vectors per tile
        const int r0 = vi0 >> 2, c0 = vi0 & 3;
        const int r1 = vi1 >> 2, c1 = vi1 & 3;
        u32x4 a0 = Ag[(size_t)(m0 + r0) * 128 + kq + c0];
        u32x4 b0 = Bg[(size_t)(n0 + r0) * 128 + kq + c0];
        u32x4 a1 = Ag[(size_t)(m0 + r1) * 128 + kq + c1];
        u32x4 b1 = Bg[(size_t)(n0 + r1) * 128 + kq + c1];
        As4[vi0] = a0; Bs4[vi0] = b0;
        As4[vi1] = a1; Bs4[vi1] = b1;
        if (k0 + 64 < DMODEL) {                         // warm L2 for chunk k+2
            __builtin_prefetch(&Ag[(size_t)(m0 + r0) * 128 + kq + 8 + c0], 0, 1);
            __builtin_prefetch(&Bg[(size_t)(n0 + r0) * 128 + kq + 8 + c0], 0, 1);
        }
    };

    stage(0, 0);
    __syncthreads();

    for (int kc = 0; kc < DMODEL / 32; ++kc) {
        const int cur = kc & 1;
        if (kc + 1 < DMODEL / 32) stage(cur ^ 1, (kc + 1) * 32);

        const u32x4* As4 = (const u32x4*)Asm[cur];      // row stride 4 u32x4
        const u32x4* Bs4 = (const u32x4*)Bsm[cur];

        ABf a[2];
#pragma unroll
        for (int mi = 0; mi < 2; ++mi) {
            const int arow = wm * 32 + mi * 16 + l15;
            a[mi].q[0] = As4[arow * 4 + hi];
            a[mi].q[1] = As4[arow * 4 + 2 + hi];
        }
#pragma unroll
        for (int ni = 0; ni < 4; ++ni) {
            ABf b;
            const int brow = wn * 64 + ni * 16 + l15;
            b.q[0] = Bs4[brow * 4 + hi];
            b.q[1] = Bs4[brow * 4 + 2 + hi];
#pragma unroll
            for (int mi = 0; mi < 2; ++mi)
                acc[mi][ni] = __builtin_amdgcn_wmma_f32_16x16x32_bf16(
                    false, a[mi].v, false, b.v, (short)0, acc[mi][ni], false, false);
        }
        __syncthreads();
    }

#pragma unroll
    for (int mi = 0; mi < 2; ++mi)
#pragma unroll
        for (int ni = 0; ni < 4; ++ni) {
            int n = n0 + wn * 64 + ni * 16 + l15;
            float badd = (bias1 ? bias1[n] : 0.f) + (bias2 ? bias2[n] : 0.f);
#pragma unroll
            for (int r = 0; r < 8; ++r) {
                int m = m0 + wm * 32 + mi * 16 + r + hi * 8;
                C[(size_t)m * ldc + n] = acc[mi][ni][r] + badd;
            }
        }
}

// ---------------------------------------------------------------------------
// Grid-wide barrier for the persistent recurrence kernel (monotonic counter)
// ---------------------------------------------------------------------------
__device__ __forceinline__ void grid_sync(unsigned* cnt, unsigned target) {
    __syncthreads();
    if (threadIdx.x == 0) {
        __hip_atomic_fetch_add(cnt, 1u, __ATOMIC_ACQ_REL, __HIP_MEMORY_SCOPE_AGENT);
        while (__hip_atomic_load(cnt, __ATOMIC_ACQUIRE, __HIP_MEMORY_SCOPE_AGENT) < target)
            __builtin_amdgcn_s_sleep(8);
    }
    __syncthreads();
}

// ---------------------------------------------------------------------------
// Persistent LSTM recurrence for one layer.
// 64 WGs x 128 threads; WG owns 16 hidden units => 64 gate rows of Wh held
// in LDS as bf16 (128 KB, b128-copied once from the pre-converted bf16
// weights). Wave w computes gate-type w for the 16 units via 32x
// v_wmma_f32_16x16x32_bf16 per step (K=1024). h ping-pongs in global.
// ---------------------------------------------------------------------------
__global__ __launch_bounds__(128)
void lstm_layer_wmma(const float* __restrict__ Xg,            // [2048,4096] incl. bx+bh
                     const unsigned* __restrict__ WhU,        // [4096,512] u32 (bf16 pairs)
                     unsigned short* __restrict__ hall,       // [2048,1024] bf16
                     unsigned short* __restrict__ hping,      // [16,1024] bf16
                     unsigned short* __restrict__ hpong,
                     float* __restrict__ hc_out,              // d_out + logits count
                     unsigned* __restrict__ counter,
                     int layer)
{
    extern __shared__ char smem[];
    unsigned* whu  = (unsigned*)smem;                   // 64 rows x 512 u32
    float*    gate = (float*)(smem + 64 * 512 * 4);     // [4][16][16]

    const int tid  = threadIdx.x;
    const int wave = tid >> 5, lane = tid & 31;
    const int l15  = lane & 15, hi = lane >> 4;
    const int U0   = blockIdx.x * 16;

    // One-time: Wh slice (rows {gt*1024 + U0 + u}) -> LDS, b128 copies
    {
        u32x4* w4 = (u32x4*)whu;
        const u32x4* Wg4 = (const u32x4*)WhU;           // row stride 128 u32x4
        for (int idx = tid; idx < 64 * 128; idx += 128) {
            int lr = idx >> 7, kk = idx & 127;
            int gt = lr >> 4, u = lr & 15;
            w4[idx] = Wg4[(size_t)(gt * DMODEL + U0 + u) * 128 + kk];
        }
    }
    __syncthreads();

    float creg = 0.f;
    const int cb = tid >> 4;    // batch 0..7
    const int cu = tid & 15;    // local unit

    for (int t = 0; t < SEQ; ++t) {
        const u32x4* hu4 = (const u32x4*)((t & 1) ? hpong : hping);  // row stride 128
        unsigned short* hn = (t & 1) ? hping : hpong;
        const u32x4* ws4 = (const u32x4*)whu;           // LDS, row stride 128

        v8f acc = {};
        for (int kq = 0; kq < 128; kq += 4) {           // kq in u32x4 units (32 bf16)
            ABf a, b;
            a.q[0] = hu4[l15 * 128 + kq + hi];
            a.q[1] = hu4[l15 * 128 + kq + 2 + hi];
            b.q[0] = ws4[(wave * 16 + l15) * 128 + kq + hi];
            b.q[1] = ws4[(wave * 16 + l15) * 128 + kq + 2 + hi];
            acc = __builtin_amdgcn_wmma_f32_16x16x32_bf16(
                false, a.v, false, b.v, (short)0, acc, false, false);
        }

        // add precomputed input gates, publish to LDS for cell update
#pragma unroll
        for (int r = 0; r < 8; ++r) {
            int m = r + hi * 8;                          // padded batch row
            float g = acc[r];
            if (m < 8)
                g += Xg[(size_t)(m * SEQ + t) * GATES + wave * DMODEL + U0 + l15];
            gate[(wave * 16 + m) * 16 + l15] = g;
        }
        __syncthreads();

        // LSTM cell: thread (cb, cu)
        float gi = gate[(0 * 16 + cb) * 16 + cu];
        float gf = gate[(1 * 16 + cb) * 16 + cu];
        float go = gate[(2 * 16 + cb) * 16 + cu];
        float gg = gate[(3 * 16 + cb) * 16 + cu];
        float si = 1.f / (1.f + expf(-gi));
        float sf = 1.f / (1.f + expf(-gf));
        float so = 1.f / (1.f + expf(-go));
        creg = sf * creg + si * tanhf(gg);
        float h = so * tanhf(creg);
        unsigned short hb = f2bf(h);
        hn[cb * DMODEL + U0 + cu] = hb;
        hall[(size_t)(cb * SEQ + t) * DMODEL + U0 + cu] = hb;
        if (t == SEQ - 1) {
            hc_out[layer * BATCH * DMODEL + cb * DMODEL + U0 + cu] = h;     // h_n
            hc_out[NLAYER * BATCH * DMODEL + layer * BATCH * DMODEL +
                   cb * DMODEL + U0 + cu] = creg;                            // c_n
        }
        grid_sync(counter, 64u * (unsigned)(t + 1));
    }
}

// ---------------------------------------------------------------------------
// RMSNorm (f32 math on bf16 h), output bf16 for the logits GEMM
// ---------------------------------------------------------------------------
__global__ __launch_bounds__(256)
void rmsnorm_kernel(const unsigned short* __restrict__ hin,
                    const float* __restrict__ nw,
                    unsigned short* __restrict__ outbf)
{
    __shared__ float red[256];
    int row = blockIdx.x;
    float vals[4], s = 0.f;
#pragma unroll
    for (int i = 0; i < 4; ++i) {
        int j = threadIdx.x + i * 256;
        float v = bf2f(hin[(size_t)row * DMODEL + j]);
        vals[i] = v; s += v * v;
    }
    red[threadIdx.x] = s; __syncthreads();
    for (int off = 128; off > 0; off >>= 1) {
        if (threadIdx.x < off) red[threadIdx.x] += red[threadIdx.x + off];
        __syncthreads();
    }
    float inv = 1.f / sqrtf(red[0] / (float)DMODEL + 1e-5f);
#pragma unroll
    for (int i = 0; i < 4; ++i) {
        int j = threadIdx.x + i * 256;
        outbf[(size_t)row * DMODEL + j] = f2bf(vals[i] * inv * nw[j]);
    }
}

// ---------------------------------------------------------------------------
extern "C" void kernel_launch(void* const* d_in, const int* in_sizes, int n_in,
                              void* d_out, int out_size, void* d_ws, size_t ws_size,
                              hipStream_t stream)
{
    (void)in_sizes; (void)n_in; (void)out_size; (void)ws_size;
    const int*   ids  = (const int*)  d_in[0];
    const float* emb  = (const float*)d_in[1];
    const float* Wx   = (const float*)d_in[2];
    const float* bx   = (const float*)d_in[3];
    const float* Wh   = (const float*)d_in[4];
    const float* bh   = (const float*)d_in[5];
    const float* nw   = (const float*)d_in[6];
    const float* outw = (const float*)d_in[7];
    float* out = (float*)d_out;

    char* ws = (char*)d_ws;
    unsigned short* xbf   = (unsigned short*)(ws + 0);                 //  4 MB
    unsigned short* hall0 = (unsigned short*)(ws + (4ull   << 20));    //  4 MB
    unsigned short* hall1 = (unsigned short*)(ws + (8ull   << 20));    //  4 MB
    unsigned short* nrm   = (unsigned short*)(ws + (12ull  << 20));    //  4 MB
    float*          Xg    = (float*)         (ws + (16ull  << 20));    // 32 MB
    unsigned short* Wxbf  = (unsigned short*)(ws + (48ull  << 20));    // 16 MB (2 layers)
    unsigned short* Whbf  = (unsigned short*)(ws + (64ull  << 20));    // 16 MB (2 layers)
    unsigned short* outwb = (unsigned short*)(ws + (80ull  << 20));    // 62.5 MB
    unsigned short* hping = (unsigned short*)(ws + (144ull << 20));    // 32 KB
    unsigned short* hpong = (unsigned short*)(ws + (144ull << 20) + 32768);
    unsigned*       cnt   = (unsigned*)      (ws + (144ull << 20) + 65536);

    const size_t logitsN = (size_t)BATCH * SEQ * VOCAB;
    const size_t WLstr   = (size_t)GATES * DMODEL;          // per-layer weight elems
    const size_t lstmLds = 64 * 512 * 4 + 4 * 16 * 16 * 4;  // 132 KB

    // one-time weight conversions to bf16
    {
        long n2;
        n2 = (long)NLAYER * WLstr / 2;
        cvt_bf16<<<(unsigned)((n2 + 255) / 256), 256, 0, stream>>>(Wx, (unsigned*)Wxbf, n2);
        cvt_bf16<<<(unsigned)((n2 + 255) / 256), 256, 0, stream>>>(Wh, (unsigned*)Whbf, n2);
        n2 = (long)VOCAB * DMODEL / 2;
        cvt_bf16<<<(unsigned)((n2 + 255) / 256), 256, 0, stream>>>(outw, (unsigned*)outwb, n2);
    }

    embed_bf16<<<MTOT, 256, 0, stream>>>(ids, emb, (unsigned*)xbf);

    // layer 0
    gemm_bf16_wmma<<<dim3(GATES / 128, MTOT / 128), 256, 0, stream>>>(
        xbf, Wxbf, bx, bh, Xg, GATES);
    hipMemsetAsync(ws + (144ull << 20), 0, 65536 + 256, stream);
    lstm_layer_wmma<<<64, 128, lstmLds, stream>>>(
        Xg, (const unsigned*)Whbf, hall0, hping, hpong, out + logitsN, cnt, 0);

    // layer 1
    gemm_bf16_wmma<<<dim3(GATES / 128, MTOT / 128), 256, 0, stream>>>(
        hall0, Wxbf + WLstr, bx + GATES, bh + GATES, Xg, GATES);
    hipMemsetAsync(ws + (144ull << 20), 0, 65536 + 256, stream);
    lstm_layer_wmma<<<64, 128, lstmLds, stream>>>(
        Xg, (const unsigned*)(Whbf + WLstr), hall1, hping, hpong, out + logitsN, cnt, 1);

    // norm + logits
    rmsnorm_kernel<<<MTOT, 256, 0, stream>>>(hall1, nw, nrm);
    gemm_bf16_wmma<<<dim3(VOCAB / 128, MTOT / 128), 256, 0, stream>>>(
        nrm, outwb, nullptr, nullptr, out, VOCAB);
}
